// FIB_RNN_40742059770157
// MI455X (gfx1250) — matrix operand order repaired and checked by reading, not compiled, not run on hardware
//
#include <hip/hip_runtime.h>
#include <hip/hip_bf16.h>
#include <math.h>

// FIB-RNN (GRU U=128, B=8192, 48 enc + 27 feedback steps) for gfx1250.
// Strategy: B-tiles of 16 rows per block; recurrent weights R^T live in
// registers as f16 WMMA B-fragments for the entire kernel; hidden state h
// lives in LDS (f32 master + f16 mirror for WMMA A). Each GRU step is
// 12 v_wmma_f32_16x16x32_f16 per wave (8 waves cover all 384 gate columns),
// followed by a fused sigmoid/tanh gate update in LDS.

typedef _Float16 half_t;
typedef __attribute__((ext_vector_type(16))) _Float16 v16h;
typedef __attribute__((ext_vector_type(8)))  _Float16 v8h;
typedef __attribute__((ext_vector_type(8)))  float    v8f;

#define BATCH    8192
#define UDIM     128
#define TN       384          // 3*U gate columns
#define MTILE    16           // batch rows per block
#define SEQ_ENC  48
#define GAMMA    28
#define NW       258          // 2*U + 2 dense-var weights
#define THREADS  256
#define NT_PER_WAVE 3         // 24 N-tiles / 8 waves
#define SOFTC    0.5413248546129181f   // log(expm1(1))

__device__ __forceinline__ float softplusf(float x) {
  return (x > 15.0f) ? x : log1pf(__expf(x));
}
__device__ __forceinline__ float sigmoidf(float x) {
  return 1.0f / (1.0f + __expf(-x));
}

__global__ __launch_bounds__(THREADS) void fib_rnn_kernel(
    const float* __restrict__ g_inputs,   // (B, 49, 1)
    const float* __restrict__ g_K,        // (1, 384)
    const float* __restrict__ g_R,        // (128, 384)
    const float* __restrict__ g_bias,     // (2, 384)
    const float* __restrict__ g_dvloc,    // (258,)
    const float* __restrict__ g_dvrho,    // (258,)
    const float* __restrict__ g_dveps,    // (28, 258)
    const float* __restrict__ g_sampeps,  // (27, B, 1)
    float* __restrict__ g_out)            // (B, 28, 2)
{
  __shared__ float  s_hg[MTILE * TN];                  // gate pre-activations (f32)
  __shared__ float  s_hf[MTILE * UDIM];                // hidden state, f32 master
  __shared__ __align__(16) half_t s_h16[MTILE * UDIM]; // hidden state, f16 mirror (WMMA A)
  __shared__ float  s_K[TN], s_b0[TN], s_b1[TN];
  __shared__ float  s_w[NW];                           // sampled dense weights this step
  __shared__ float  s_x[MTILE];                        // per-row scalar GRU input
  __shared__ float  s_x2[MTILE * 2];                   // dense head output

  const int tid   = threadIdx.x;
  const int lane  = tid & 31;
  const int wave  = tid >> 5;
  const int bbase = blockIdx.x * MTILE;

  for (int i = tid; i < TN; i += THREADS) {
    s_K[i]  = g_K[i];
    s_b0[i] = g_bias[i];
    s_b1[i] = g_bias[TN + i];
  }
  for (int i = tid; i < MTILE * UDIM; i += THREADS) {
    s_hf[i]  = 0.0f;
    s_h16[i] = (half_t)0.0f;
  }

  // ---- Load resident B fragments: R^T as f16, 32x16 KxN WMMA layout.
  // Lane l holds column n = ntile*16 + (l&15); elems 0..7 = K kb*32 + (l>>4)*8 + i, elems 8..15 = K kb*32 + 16 + (l>>4)*8 + (i-8).
  v16h Bfrag[NT_PER_WAVE][4];
  {
    const int nloc  = lane & 15;
    const int khalf = (lane >> 4) * 8;
    for (int j = 0; j < NT_PER_WAVE; ++j) {
      const int n = (wave * NT_PER_WAVE + j) * 16 + nloc;
      for (int kb = 0; kb < 4; ++kb) {
        const int kbase = kb * 32 + khalf;
        v16h f;
        #pragma unroll
        for (int i = 0; i < 16; ++i)
          f[i] = (half_t)g_R[(kbase + i + (i >= 8 ? 8 : 0)) * TN + n];
        Bfrag[j][kb] = f;
      }
    }
  }
  __syncthreads();

  // ---- One GRU step: hg = h @ R (WMMA), then fused gates. s_x must be set.
  auto gru_step = [&]() {
    {
      // A fragments (16x32 f16): lane l holds row (l&15);
      // elems 0..7 = K base..base+7, elems 8..15 = K base+16..base+23,
      // base = kb*32 + (l>>4)*8. Two 16B LDS loads per fragment.
      const int row  = lane & 15;
      const int koff = (lane >> 4) * 8;
      v16h A[4];
      #pragma unroll
      for (int kb = 0; kb < 4; ++kb) {
        const v8h* p = reinterpret_cast<const v8h*>(&s_h16[row * UDIM + kb * 32 + koff]);
        union { v16h v; v8h h[2]; } u;
        u.h[0] = p[0];   // K = base .. base+7
        u.h[1] = p[2];   // K = base+16 .. base+23
        A[kb] = u.v;
      }
      #pragma unroll
      for (int j = 0; j < NT_PER_WAVE; ++j) {
        v8f c = {};
        #pragma unroll
        for (int kb = 0; kb < 4; ++kb)
          c = __builtin_amdgcn_wmma_f32_16x16x32_f16(
                false, A[kb], false, Bfrag[j][kb], (short)0, c, false, false);
        // C layout: VGPR i -> row i + 8*(lane>=16), col = lane&15
        const int nbase = (wave * NT_PER_WAVE + j) * 16 + (lane & 15);
        const int rbase = (lane >> 4) * 8;
        #pragma unroll
        for (int i = 0; i < 8; ++i)
          s_hg[(rbase + i) * TN + nbase] = c[i];
      }
    }
    __syncthreads();
    // Fused gate update: z, r, candidate, new h. Folds xg = x*K + b0.
    for (int e = tid; e < MTILE * UDIM; e += THREADS) {
      const int row = e >> 7;           // U == 128
      const int j   = e & (UDIM - 1);
      const float x  = s_x[row];
      const float hz = s_hg[row * TN + j]            + s_b1[j];
      const float hr = s_hg[row * TN + UDIM + j]     + s_b1[UDIM + j];
      const float hc = s_hg[row * TN + 2 * UDIM + j] + s_b1[2 * UDIM + j];
      const float xz = fmaf(x, s_K[j],            s_b0[j]);
      const float xr = fmaf(x, s_K[UDIM + j],     s_b0[UDIM + j]);
      const float xh = fmaf(x, s_K[2 * UDIM + j], s_b0[2 * UDIM + j]);
      const float z  = sigmoidf(xz + hz);
      const float r  = sigmoidf(xr + hr);
      const float hh = tanhf(fmaf(r, hc, xh));
      const float hn = z * s_hf[e] + (1.0f - z) * hh;
      s_hf[e]  = hn;
      s_h16[e] = (half_t)hn;
    }
    __syncthreads();
  };

  // ---- Stochastic dense head for step g: x2 = h @ w(g) + b(g); emit output.
  auto dense_out = [&](int g) {
    for (int i = tid; i < NW; i += THREADS) {
      const float sc = 1e-5f + 0.02f * softplusf(SOFTC + g_dvrho[i]);
      s_w[i] = fmaf(sc, g_dveps[g * NW + i], g_dvloc[i]);
    }
    __syncthreads();
    if (tid < MTILE * 2) {
      const int row = tid >> 1;
      const int c   = tid & 1;
      float acc = s_w[2 * UDIM + c];
      #pragma unroll 8
      for (int k = 0; k < UDIM; ++k)
        acc = fmaf(s_hf[row * UDIM + k], s_w[2 * k + c], acc);
      s_x2[row * 2 + c] = acc;
      const float o = (c == 0) ? acc
                               : (1e-5f + 0.05f * softplusf(SOFTC + acc));
      g_out[(size_t)(bbase + row) * (GAMMA * 2) + g * 2 + c] = o;
    }
    __syncthreads();
  };

  // ---- Encoder: 48 steps driven by the input series.
  for (int t = 0; t < SEQ_ENC; ++t) {
    if (tid < MTILE)
      s_x[tid] = g_inputs[(size_t)(bbase + tid) * 49 + t];
    __syncthreads();
    gru_step();
  }
  dense_out(0);

  // ---- Feedback: 27 steps driven by sampled predictions.
  for (int g = 1; g < GAMMA; ++g) {
    if (tid < MTILE) {
      const float loc = s_x2[tid * 2];
      const float sc  = 1e-5f + 0.05f * softplusf(SOFTC + s_x2[tid * 2 + 1]);
      s_x[tid] = fmaf(sc, g_sampeps[(size_t)(g - 1) * BATCH + bbase + tid], loc);
    }
    __syncthreads();
    gru_step();
    dense_out(g);
  }
}

extern "C" void kernel_launch(void* const* d_in, const int* in_sizes, int n_in,
                              void* d_out, int out_size, void* d_ws, size_t ws_size,
                              hipStream_t stream) {
  (void)in_sizes; (void)n_in; (void)out_size; (void)d_ws; (void)ws_size;
  const float* inputs  = (const float*)d_in[0];
  const float* K       = (const float*)d_in[1];
  const float* R       = (const float*)d_in[2];
  const float* bias    = (const float*)d_in[3];
  const float* dvloc   = (const float*)d_in[4];
  const float* dvrho   = (const float*)d_in[5];
  const float* dveps   = (const float*)d_in[6];
  const float* sampeps = (const float*)d_in[7];
  float* out = (float*)d_out;

  dim3 grid(BATCH / MTILE);   // 512 blocks
  dim3 block(THREADS);        // 8 wave32
  hipLaunchKernelGGL(fib_rnn_kernel, grid, block, 0, stream,
                     inputs, K, R, bias, dvloc, dvrho, dveps, sampeps, out);
}
